// KPConvFPN_22643067584625
// MI455X (gfx1250) — matrix operand order repaired and checked by reading, not compiled, not run on hardware
//
#include <hip/hip_runtime.h>

// ============================================================================
// KPConv-FPN forward for MI455X (gfx1250, wave32, WMMA).
//
// All matmuls (unary layers, kpconv weight application, shortcut, decoder)
// go through one bf16 WMMA GEMM (v_wmma_f32_16x16x32_bf16, fp32 accumulate).
// Weights are converted f32->bf16 *transposed* (Bt = W^T, n-major) so both A
// and B WMMA fragments load from LDS as 16-byte ds_load_b128 chunks.
// The KPConv gather kernel produces the [N, P*Cin] weighted-feature matrix
// directly in bf16, halving the bytes of the biggest intermediate.
//
// Input pointer layout (JAX pytree flatten; dicts sorted by key):
//   d_in[0..13]: feats, points0..3, neigh0..3, sub0..2, up1, up2
//   d_in[14..]:  conv0{b,g,kpoints,kw}, dec0{W,b,g}, dec1{W,bias},
//                rbs[i]{kpoints,kw,nb,ng,[sc.{W,b,g}],[u1.{W,b,g}],u2.{W,b,g}}
// ============================================================================

typedef __bf16 bf16;
typedef __attribute__((ext_vector_type(16))) __bf16 v16bf;
typedef __attribute__((ext_vector_type(8)))  __bf16 v8bf;
typedef __attribute__((ext_vector_type(4)))  __bf16 v4bf;
typedef __attribute__((ext_vector_type(8)))  float  v8f;

#define KNB 40     // neighbors
#define PKP 15     // kernel points
#define NGRP 32    // group-norm groups

#define LEAKY(t) ((t) > 0.f ? (t) : 0.1f * (t))

// ------------------------------------------------------------------ elementwise
__global__ __launch_bounds__(256) void kfpn_cvt_bf16(const float* __restrict__ s,
                                                     bf16* __restrict__ d, long n) {
  long i = (long)blockIdx.x * 256 + threadIdx.x;
  if (i < n) d[i] = (bf16)s[i];
}

// transpose-convert: dst[n*ldd + k] = (bf16) src[k*N + n]   (src is [K][N] f32)
__global__ __launch_bounds__(256) void kfpn_cvt_bf16_t(const float* __restrict__ src,
                                                       bf16* __restrict__ dst,
                                                       int K, int N, int ldd) {
  long i = (long)blockIdx.x * 256 + threadIdx.x;
  if (i >= (long)N * K) return;
  int n = (int)(i / K), k = (int)(i % K);
  dst[(long)n * ldd + k] = (bf16)src[(long)k * N + n];
}

__global__ __launch_bounds__(256) void kfpn_div_cnt(float* __restrict__ x,
                                                    const float* __restrict__ cnt,
                                                    long Nr, int C) {
  long i = (long)blockIdx.x * 256 + threadIdx.x;
  if (i < Nr * (long)C) x[i] = x[i] / cnt[i / C];
}

__global__ __launch_bounds__(256) void kfpn_add_leaky(const float* __restrict__ a,
                                                      const float* __restrict__ b,
                                                      float* __restrict__ y, long n) {
  long i = (long)blockIdx.x * 256 + threadIdx.x;
  if (i < n) { float t = a[i] + b[i]; y[i] = LEAKY(t); }
}

__global__ __launch_bounds__(256) void kfpn_bias_add(float* __restrict__ y,
                                                     const float* __restrict__ bias,
                                                     long Nr, int C) {
  long i = (long)blockIdx.x * 256 + threadIdx.x;
  if (i < Nr * (long)C) y[i] += bias[i % C];
}

__global__ __launch_bounds__(256) void kfpn_maxpool(const float* __restrict__ sf, int C, int Ms,
                                                    const int* __restrict__ neigh,
                                                    float* __restrict__ out, long Nq) {
  long i = (long)blockIdx.x * 256 + threadIdx.x;
  if (i >= Nq * (long)C) return;
  long n = i / C; int c = (int)(i % C);
  float m = -3.402823466e38f;
  for (int k = 0; k < KNB; ++k) {
    int idx = neigh[n * KNB + k];
    float v = (idx < Ms) ? sf[(long)idx * C + c] : 0.f;
    m = fmaxf(m, v);
  }
  out[i] = m;
}

// nearest upsample: dst[n, col0 + c] = x[idx[n*K + 0], c]
__global__ __launch_bounds__(256) void kfpn_gather_row0(const float* __restrict__ x, int C, int Ms,
                                                        const int* __restrict__ idx,
                                                        float* __restrict__ dst,
                                                        int ldd, int col0, long Nq) {
  long i = (long)blockIdx.x * 256 + threadIdx.x;
  if (i >= Nq * (long)C) return;
  long n = i / C; int c = (int)(i % C);
  int r = idx[n * KNB];
  dst[n * (long)ldd + col0 + c] = (r < Ms) ? x[(long)r * C + c] : 0.f;
}

__global__ __launch_bounds__(256) void kfpn_copy_cols(const float* __restrict__ src, int C,
                                                      float* __restrict__ dst,
                                                      int ldd, int col0, long Nr) {
  long i = (long)blockIdx.x * 256 + threadIdx.x;
  if (i >= Nr * (long)C) return;
  long n = i / C; int c = (int)(i % C);
  dst[n * (long)ldd + col0 + c] = src[i];
}

// ---------------------------------------------------------------- group norm
// pack-mode GroupNorm over [Nr, C]: group g covers channels [g*C/G, (g+1)*C/G)
__global__ __launch_bounds__(256) void kfpn_gn_stats(const float* __restrict__ x,
                                                     long Nr, int C,
                                                     float* __restrict__ stats) {
  int g = blockIdx.x;
  int cg = C / NGRP;
  long total = Nr * (long)cg;
  float s = 0.f, ss = 0.f;
  for (long i = threadIdx.x; i < total; i += 256) {
    long r = i / cg; int c = g * cg + (int)(i % cg);
    float v = x[r * (long)C + c];
    s += v; ss += v * v;
  }
  __shared__ float rs[256], rq[256];
  rs[threadIdx.x] = s; rq[threadIdx.x] = ss;
  __syncthreads();
  for (int o = 128; o > 0; o >>= 1) {
    if ((int)threadIdx.x < o) {
      rs[threadIdx.x] += rs[threadIdx.x + o];
      rq[threadIdx.x] += rq[threadIdx.x + o];
    }
    __syncthreads();
  }
  if (threadIdx.x == 0) {
    float inv = 1.f / (float)total;
    float m = rs[0] * inv;
    float v = rq[0] * inv - m * m;
    stats[2 * g] = m;
    stats[2 * g + 1] = v;
  }
}

__global__ __launch_bounds__(256) void kfpn_gn_apply(const float* __restrict__ x,
                                                     float* __restrict__ y,
                                                     long Nr, int C,
                                                     const float* __restrict__ gamma,
                                                     const float* __restrict__ beta,
                                                     const float* __restrict__ stats, int relu) {
  long i = (long)blockIdx.x * 256 + threadIdx.x;
  if (i >= Nr * (long)C) return;
  int c = (int)(i % C); int g = c / (C / NGRP);
  float m = stats[2 * g], v = stats[2 * g + 1];
  float t = (x[i] - m) * rsqrtf(v + 1e-5f) * gamma[c] + beta[c];
  y[i] = relu ? LEAKY(t) : t;
}

// -------------------------------------------------------------- kpconv gather
// One block per query point. Builds wf[n*ldwf + p*Cin + c] = sum_k infl[k,p]*sf[nb,c]
// in bf16 (A matrix for the WMMA GEMM against kw[(p,c), d]), plus valid count.
__global__ __launch_bounds__(256) void kfpn_kpconv_gather(
    const float* __restrict__ sf, int Cin, int Ms,
    const float* __restrict__ qp, const float* __restrict__ sp,
    const int* __restrict__ neigh,
    const float* __restrict__ kpts, float sigma,
    bf16* __restrict__ wf, int ldwf, float* __restrict__ cnt, int Nq) {
  int n = blockIdx.x;
  if (n >= Nq) return;
  __shared__ float s_infl[KNB * PKP];
  __shared__ int   s_idx[KNB];
  __shared__ float s_q[3];
  __shared__ int   s_cnt;
  int tid = threadIdx.x;
  if (tid == 0) s_cnt = 0;
  if (tid < 3) s_q[tid] = qp[(long)n * 3 + tid];
  if (tid < KNB) s_idx[tid] = neigh[(long)n * KNB + tid];
  __syncthreads();
  if (tid < KNB * PKP) {
    int k = tid / PKP, p = tid % PKP;
    int idx = s_idx[k];
    float ox, oy, oz;
    if (idx < Ms) {
      ox = sp[(long)idx * 3 + 0] - s_q[0];
      oy = sp[(long)idx * 3 + 1] - s_q[1];
      oz = sp[(long)idx * 3 + 2] - s_q[2];
    } else {  // padded far point
      ox = 1e6f - s_q[0]; oy = 1e6f - s_q[1]; oz = 1e6f - s_q[2];
    }
    float dx = ox - kpts[p * 3 + 0];
    float dy = oy - kpts[p * 3 + 1];
    float dz = oz - kpts[p * 3 + 2];
    float d = sqrtf(dx * dx + dy * dy + dz * dz);
    s_infl[tid] = fmaxf(1.f - d / sigma, 0.f);
  }
  if (tid < KNB) {
    int idx = s_idx[tid];
    float s = 0.f;
    if (idx < Ms) {
      const float* row = sf + (long)idx * Cin;
      for (int c = 0; c < Cin; ++c) s += row[c];
    }
    if (s > 0.f) atomicAdd(&s_cnt, 1);
  }
  __syncthreads();
  for (int c = tid; c < Cin; c += 256) {
    float acc[PKP];
#pragma unroll
    for (int p = 0; p < PKP; ++p) acc[p] = 0.f;
    for (int k = 0; k < KNB; ++k) {
      int idx = s_idx[k];
      float f = (idx < Ms) ? sf[(long)idx * Cin + c] : 0.f;
      const float* ip = &s_infl[k * PKP];
#pragma unroll
      for (int p = 0; p < PKP; ++p) acc[p] += ip[p] * f;
    }
    long base = (long)n * ldwf;
#pragma unroll
    for (int p = 0; p < PKP; ++p) wf[base + (long)p * Cin + c] = (bf16)acc[p];
  }
  if (tid == 0) { int c = s_cnt; cnt[n] = (float)(c < 1 ? 1 : c); }
}

// ------------------------------------------------------------------ WMMA GEMM
// C[M,N] = A[M,K] * B[K,N]; A bf16 [M,K] row-major, Bt bf16 [N,K] row-major
// (i.e. B transposed), C fp32. Block: 256 threads = 8 waves, tile 128(M) x
// 64(N), K-step 32. Waves 4(M) x 2(N); each wave: 32x32 = 2x2 wmma accums.
// LDS rows padded to 40 halves (80 B): keeps 16-byte alignment for b128 loads.
__global__ __launch_bounds__(256) void kfpn_gemm_bf16_wmma(
    const bf16* __restrict__ A, int lda,
    const bf16* __restrict__ Bt, int ldbt,
    float* __restrict__ C, int ldc,
    int M, int N, int Kd) {
  __shared__ bf16 As[128][40];   // [m][k]
  __shared__ bf16 Bs[64][40];    // [n][k]
  int tid  = threadIdx.x;
  int wave = tid >> 5, lane = tid & 31;
  int wm = (wave >> 1) * 32;
  int wn = (wave & 1) * 32;
  int bm = blockIdx.x * 128;
  int bn = blockIdx.y * 64;

  v8f acc[2][2] = {};

  for (int k0 = 0; k0 < Kd; k0 += 32) {
    // prefetch next K tile (global_prefetch_b8)
    if (k0 + 32 < Kd) {
      if (tid < 128) {
        int gr = bm + tid;
        if (gr < M) __builtin_prefetch(A + (long)gr * lda + k0 + 32, 0, 1);
      } else if (tid < 192) {
        int gn = bn + (tid - 128);
        if (gn < N) __builtin_prefetch(Bt + (long)gn * ldbt + k0 + 32, 0, 1);
      }
    }
    // stage A tile 128x32 in 4-bf16 (8 B) chunks: 1024 chunks, 4 per thread
#pragma unroll
    for (int it = 0; it < 4; ++it) {
      int ch = tid + it * 256;
      int r  = ch >> 3;
      int kk = (ch & 7) * 4;
      int gr = bm + r, gk = k0 + kk;
      if (gr < M && gk + 4 <= Kd) {
        v4bf w = *(const v4bf*)(A + (long)gr * lda + gk);
        *(v4bf*)&As[r][kk] = w;
      } else {
#pragma unroll
        for (int e = 0; e < 4; ++e) {
          bf16 v = (bf16)0.0f;
          if (gr < M && gk + e < Kd) v = A[(long)gr * lda + gk + e];
          As[r][kk + e] = v;
        }
      }
    }
    // stage Bt tile 64x32 in 4-bf16 chunks: 512 chunks, 2 per thread
#pragma unroll
    for (int it = 0; it < 2; ++it) {
      int ch = tid + it * 256;
      int r  = ch >> 3;
      int kk = (ch & 7) * 4;
      int gn = bn + r, gk = k0 + kk;
      if (gn < N && gk + 4 <= Kd) {
        v4bf w = *(const v4bf*)(Bt + (long)gn * ldbt + gk);
        *(v4bf*)&Bs[r][kk] = w;
      } else {
#pragma unroll
        for (int e = 0; e < 4; ++e) {
          bf16 v = (bf16)0.0f;
          if (gn < N && gk + e < Kd) v = Bt[(long)gn * ldbt + gk + e];
          Bs[r][kk + e] = v;
        }
      }
    }
    __syncthreads();

    // A fragment (16x32): lanes 0-15 -> K 0-7 & 16-23; lanes 16-31 -> K 8-15 & 24-31
    int arow = lane & 15, kb = (lane >> 4) * 8;
    v16bf afrag[2], bfrag[2];
#pragma unroll
    for (int i = 0; i < 2; ++i) {
      int r = wm + i * 16 + arow;
      v8bf lo = *(const v8bf*)&As[r][kb];
      v8bf hi = *(const v8bf*)&As[r][kb + 16];
      afrag[i] = __builtin_shufflevector(lo, hi, 0, 1, 2, 3, 4, 5, 6, 7,
                                         8, 9, 10, 11, 12, 13, 14, 15);
    }
    // B fragment (32x16): lane n = lane&15; lanes 0-15 -> K 0-7 & 16-23; lanes 16-31 -> K 8-15 & 24-31
    int bcol = lane & 15, kbb = (lane >> 4) * 8;
#pragma unroll
    for (int i = 0; i < 2; ++i) {
      int cc = wn + i * 16 + bcol;
      v8bf lo = *(const v8bf*)&Bs[cc][kbb];
      v8bf hi = *(const v8bf*)&Bs[cc][kbb + 16];
      bfrag[i] = __builtin_shufflevector(lo, hi, 0, 1, 2, 3, 4, 5, 6, 7,
                                         8, 9, 10, 11, 12, 13, 14, 15);
    }
#pragma unroll
    for (int i = 0; i < 2; ++i)
#pragma unroll
      for (int j = 0; j < 2; ++j)
        acc[i][j] = __builtin_amdgcn_wmma_f32_16x16x32_bf16(
            false, afrag[i], false, bfrag[j], (short)0, acc[i][j], false, false);
    __syncthreads();
  }

  // store: VGPR r -> row (lane<16 ? r : r+8), col = lane&15
  int crow0 = (lane < 16) ? 0 : 8;
  int ccol  = lane & 15;
#pragma unroll
  for (int i = 0; i < 2; ++i)
#pragma unroll
    for (int j = 0; j < 2; ++j)
#pragma unroll
      for (int r = 0; r < 8; ++r) {
        int gr = bm + wm + i * 16 + crow0 + r;
        int gc = bn + wn + j * 16 + ccol;
        if (gr < M && gc < N) C[(long)gr * ldc + gc] = acc[i][j][r];
      }
}

// ================================ host side =================================
static inline dim3 grid1(long n) { return dim3((unsigned)((n + 255) / 256)); }

struct KScratch {
  bf16 *wbuf, *abuf, *wf;
  float *cnt, *stats;
  float *x0, *x1, *tA, *tB, *tC, *tD, *tE;
  float *res1, *res2, *res3, *cat2, *l2, *l1cat;
};

static void h_cvt(hipStream_t s, const float* src, bf16* dst, long n) {
  kfpn_cvt_bf16<<<grid1(n), 256, 0, s>>>(src, dst, n);
}
// W [K][N] f32 -> Wt [N][ldd] bf16
static void h_cvt_t(hipStream_t s, const float* W, bf16* Wt, int K, int N, int ldd) {
  kfpn_cvt_bf16_t<<<grid1((long)K * N), 256, 0, s>>>(W, Wt, K, N, ldd);
}
static void h_gemm(hipStream_t s, const bf16* A, int lda, const bf16* Bt, int ldbt,
                   float* C, int ldc, int M, int N, int Kd) {
  dim3 g((unsigned)((M + 127) / 128), (unsigned)((N + 63) / 64));
  kfpn_gemm_bf16_wmma<<<g, 256, 0, s>>>(A, lda, Bt, ldbt, C, ldc, M, N, Kd);
}
static void h_gnorm(hipStream_t s, const float* x, float* y, long Nr, int C,
                    const float* g, const float* b, float* stats, int relu) {
  kfpn_gn_stats<<<NGRP, 256, 0, s>>>(x, Nr, C, stats);
  kfpn_gn_apply<<<grid1(Nr * (long)C), 256, 0, s>>>(x, y, Nr, C, g, b, stats, relu);
}
// y = gnorm(x @ W, g, b) [+ leaky]
static void h_unary(hipStream_t s, KScratch& S, const float* x, long Nr, int ci,
                    const float* W, const float* gg, const float* bb,
                    float* y, int co, int relu) {
  h_cvt(s, x, S.abuf, Nr * (long)ci);
  h_cvt_t(s, W, S.wbuf, ci, co, ci);
  h_gemm(s, S.abuf, ci, S.wbuf, ci, y, co, (int)Nr, co, ci);
  h_gnorm(s, y, y, Nr, co, gg, bb, S.stats, relu);
}

extern "C" void kernel_launch(void* const* d_in, const int* in_sizes, int n_in,
                              void* d_out, int out_size, void* d_ws, size_t ws_size,
                              hipStream_t stream) {
  (void)in_sizes; (void)out_size; (void)ws_size;
  if (n_in < 135) return;

  const int N0 = 40000, N1 = 10000, N2 = 2500, N3 = 800;

  const float* feats  = (const float*)d_in[0];
  const float* p0     = (const float*)d_in[1];
  const float* p1     = (const float*)d_in[2];
  const float* p2     = (const float*)d_in[3];
  const float* p3     = (const float*)d_in[4];
  const int*   neigh0 = (const int*)d_in[5];
  const int*   neigh1 = (const int*)d_in[6];
  const int*   neigh2 = (const int*)d_in[7];
  const int*   neigh3 = (const int*)d_in[8];
  const int*   sub0   = (const int*)d_in[9];
  const int*   sub1   = (const int*)d_in[10];
  const int*   sub2   = (const int*)d_in[11];
  const int*   up1    = (const int*)d_in[12];
  const int*   up2    = (const int*)d_in[13];

  int ti = 14;
  auto F = [&]() { return (const float*)d_in[ti++]; };

  const float *c0_b = F(), *c0_g = F(), *c0_kp = F(), *c0_kw = F();
  const float *dec0_W = F(), *dec0_b = F(), *dec0_g = F();
  const float *dec1_W = F(), *dec1_bias = F();

  struct RBP { const float *kp,*kw,*nb,*ng,*scW,*scb,*scg,*u1W,*u1b,*u1g,*u2W,*u2b,*u2g; };
  const int CIN[10]  = {64,128,128,256,256,256,512,512,512,1024};
  const int COUT[10] = {128,128,256,256,256,512,512,512,1024,1024};
  const float SIG[10]= {0.6f,0.6f,1.2f,1.2f,1.2f,2.4f,2.4f,2.4f,4.8f,4.8f};
  const int STRIDED[10] = {0,1,0,0,1,0,0,1,0,0};
  RBP rb[10];
  for (int i = 0; i < 10; ++i) {
    int mid = COUT[i] / 4;
    bool has_sc = (CIN[i] != COUT[i]);
    bool has_u1 = (CIN[i] != mid);
    rb[i].kp = F(); rb[i].kw = F(); rb[i].nb = F(); rb[i].ng = F();
    if (has_sc) { rb[i].scW = F(); rb[i].scb = F(); rb[i].scg = F(); }
    else        { rb[i].scW = rb[i].scb = rb[i].scg = nullptr; }
    if (has_u1) { rb[i].u1W = F(); rb[i].u1b = F(); rb[i].u1g = F(); }
    else        { rb[i].u1W = rb[i].u1b = rb[i].u1g = nullptr; }
    rb[i].u2W = F(); rb[i].u2b = F(); rb[i].u2g = F();
  }

  // ---- scratch carve-out (deterministic bump allocator) ----
  char* base = (char*)d_ws; size_t off = 0;
  auto salloc = [&](size_t bytes) -> void* {
    off = (off + 255) & ~(size_t)255;
    void* p = base + off; off += bytes; return p;
  };
  KScratch S;
  S.wbuf  = (bf16*)salloc((size_t)983040 * 2);     // max weight (rb8/9 kw)
  S.abuf  = (bf16*)salloc((size_t)7680000 * 2);    // max GEMM A staging (l1cat)
  S.wf    = (bf16*)salloc((size_t)19200000 * 2);   // max wf: 40000 x 480
  S.cnt   = (float*)salloc((size_t)40000 * 4);
  S.stats = (float*)salloc(64 * 4);
  S.x0    = (float*)salloc((size_t)5120000 * 4);
  S.x1    = (float*)salloc((size_t)5120000 * 4);
  S.tA    = (float*)salloc((size_t)1280000 * 4);   // unary1 out [Ns, mid]
  S.tB    = (float*)salloc((size_t)1280000 * 4);   // kpconv out [Nq, mid]
  S.tC    = (float*)salloc((size_t)5120000 * 4);   // unary2 out [Nq, cout]
  S.tD    = (float*)salloc((size_t)1280000 * 4);   // maxpool out [Nq, cin]
  S.tE    = (float*)salloc((size_t)5120000 * 4);   // shortcut out [Nq, cout]
  S.res1  = (float*)salloc((size_t)N1 * 256 * 4);
  S.res2  = (float*)salloc((size_t)N2 * 512 * 4);
  S.res3  = (float*)salloc((size_t)N3 * 1024 * 4);
  S.cat2  = (float*)salloc((size_t)N2 * 1536 * 4);
  S.l2    = (float*)salloc((size_t)N2 * 512 * 4);
  S.l1cat = (float*)salloc((size_t)N1 * 768 * 4);

  // ---- conv0: kpconv(feats) -> /cnt -> gnorm+leaky ----
  // wf leading dim padded 15 -> 16 so GEMM A rows stay 8-byte aligned.
  kfpn_kpconv_gather<<<N0, 256, 0, stream>>>(feats, 1, N0, p0, p0, neigh0,
                                             c0_kp, 0.6f, S.wf, 16, S.cnt, N0);
  h_cvt_t(stream, c0_kw, S.wbuf, PKP, 64, 16);     // [15][64] -> [64][16(pad)]
  h_gemm(stream, S.wf, 16, S.wbuf, 16, S.x0, 64, N0, 64, PKP);
  kfpn_div_cnt<<<grid1((long)N0 * 64), 256, 0, stream>>>(S.x0, S.cnt, N0, 64);
  h_gnorm(stream, S.x0, S.x0, N0, 64, c0_g, c0_b, S.stats, 1);

  // ---- residual blocks ----
  const float* QP[10] = {p0, p1, p1, p1, p2, p2, p2, p3, p3, p3};
  const float* SP[10] = {p0, p0, p1, p1, p1, p2, p2, p2, p3, p3};
  const int*   NGH[10]= {neigh0, sub0, neigh1, neigh1, sub1, neigh2, neigh2, sub2, neigh3, neigh3};
  const int NQ[10] = {N0, N1, N1, N1, N2, N2, N2, N3, N3, N3};
  const int NS[10] = {N0, N0, N1, N1, N1, N2, N2, N2, N3, N3};
  float* OUT[10] = {S.x1, S.x0, S.x1, S.res1, S.x0, S.x1, S.res2, S.x0, S.x1, S.res3};

  const float* x = S.x0;
  for (int i = 0; i < 10; ++i) {
    int cin = CIN[i], cout = COUT[i], mid = cout / 4;
    long Nq = NQ[i], Ns = NS[i];
    // unary1
    const float* midfeat = x;
    if (rb[i].u1W) {
      h_unary(stream, S, x, Ns, cin, rb[i].u1W, rb[i].u1g, rb[i].u1b, S.tA, mid, 1);
      midfeat = S.tA;
    }
    // kpconv: gather -> WMMA GEMM against kw[(p,c), d] -> /cnt -> gnorm+leaky
    int Kk = PKP * mid;
    kfpn_kpconv_gather<<<(unsigned)Nq, 256, 0, stream>>>(midfeat, mid, (int)Ns,
        QP[i], SP[i], NGH[i], rb[i].kp, SIG[i], S.wf, Kk, S.cnt, (int)Nq);
    h_cvt_t(stream, rb[i].kw, S.wbuf, Kk, mid, Kk);
    h_gemm(stream, S.wf, Kk, S.wbuf, Kk, S.tB, mid, (int)Nq, mid, Kk);
    kfpn_div_cnt<<<grid1(Nq * (long)mid), 256, 0, stream>>>(S.tB, S.cnt, Nq, mid);
    h_gnorm(stream, S.tB, S.tB, Nq, mid, rb[i].ng, rb[i].nb, S.stats, 1);
    // unary2 (no relu)
    h_unary(stream, S, S.tB, Nq, mid, rb[i].u2W, rb[i].u2g, rb[i].u2b, S.tC, cout, 0);
    // shortcut
    const float* sc = x;
    if (STRIDED[i]) {
      kfpn_maxpool<<<grid1(Nq * (long)cin), 256, 0, stream>>>(x, cin, (int)Ns, NGH[i], S.tD, Nq);
      sc = S.tD;
    }
    const float* sh = sc;
    if (rb[i].scW) {
      h_unary(stream, S, sc, Nq, cin, rb[i].scW, rb[i].scg, rb[i].scb, S.tE, cout, 0);
      sh = S.tE;
    }
    kfpn_add_leaky<<<grid1(Nq * (long)cout), 256, 0, stream>>>(S.tC, sh, OUT[i], Nq * (long)cout);
    x = OUT[i];
  }

  // ---- decoder ----
  // cat2 = [upsample(res3, up2) | res2] : [N2, 1536]
  kfpn_gather_row0<<<grid1((long)N2 * 1024), 256, 0, stream>>>(S.res3, 1024, N3, up2,
                                                              S.cat2, 1536, 0, N2);
  kfpn_copy_cols<<<grid1((long)N2 * 512), 256, 0, stream>>>(S.res2, 512, S.cat2, 1536, 1024, N2);
  h_unary(stream, S, S.cat2, N2, 1536, dec0_W, dec0_g, dec0_b, S.l2, 512, 1);
  // l1cat = [upsample(l2, up1) | res1] : [N1, 768]
  kfpn_gather_row0<<<grid1((long)N1 * 512), 256, 0, stream>>>(S.l2, 512, N2, up1,
                                                              S.l1cat, 768, 0, N1);
  kfpn_copy_cols<<<grid1((long)N1 * 256), 256, 0, stream>>>(S.res1, 256, S.l1cat, 768, 512, N1);
  // out = l1cat @ dec1.W + bias
  h_cvt(stream, S.l1cat, S.abuf, (long)N1 * 768);
  h_cvt_t(stream, dec1_W, S.wbuf, 768, 256, 768);
  h_gemm(stream, S.abuf, 768, S.wbuf, 768, (float*)d_out, 256, N1, 256, 768);
  kfpn_bias_add<<<grid1((long)N1 * 256), 256, 0, stream>>>((float*)d_out, dec1_bias, N1, 256);
}
